// StructureModule_70420283785941
// MI455X (gfx1250) — compile-verified, matrix operand order checked by reading, not run on hardware
//
#include <hip/hip_runtime.h>
#include <math.h>

// ---------------------------------------------------------------------------
// CDNA5 (gfx1250) wave32 WMMA types & helpers
// ---------------------------------------------------------------------------
typedef __attribute__((ext_vector_type(16))) __bf16 v16bf;
typedef __attribute__((ext_vector_type(8)))  float  v8f;
typedef __attribute__((ext_vector_type(4)))  unsigned int tdm_v4u;
typedef __attribute__((ext_vector_type(8)))  int          tdm_v8i;
typedef __attribute__((ext_vector_type(4)))  int          tdm_v4i;

#define NRES 512
#define CS_  384
#define CZ_  128
#define H_   12
#define CH_  16
#define C_   128
#define NT_  7
#define WL_  0.57735026918962576f   /* sqrt(1/3)  */
#define WC_  0.23570226039551587f   /* sqrt(2/36) */

__device__ __forceinline__ __bf16 f2bf(float f){
  union { float f; unsigned u; } a; a.f = f;
  unsigned r = a.u + 0x7FFFu + ((a.u >> 16) & 1u);   // round-to-nearest-even
  union { unsigned short s; __bf16 b; } o; o.s = (unsigned short)(r >> 16);
  return o.b;
}
__device__ __forceinline__ unsigned short f2bfu(float f){
  union { float f; unsigned u; } a; a.f = f;
  unsigned r = a.u + 0x7FFFu + ((a.u >> 16) & 1u);
  return (unsigned short)(r >> 16);
}

// K-index of half element e (0..15) for lane-group g (0/1), per ISA 16-bit
// A-matrix 16x32 layout (VGPR0: K=0,1 | 8,9 ; ... VGPR4-7: +16).
__device__ __forceinline__ int kmap(int e, int g){
  int v = e >> 1, p = e & 1;
  return (v < 4) ? (2*v + p + 8*g) : (16 + 2*(v-4) + p + 8*g);
}

__device__ __forceinline__ v8f wmma_bf16(v16bf a, v16bf b, v8f c){
  return __builtin_amdgcn_wmma_f32_16x16x32_bf16(false, a, false, b, (short)0, c,
                                                 false, false);
}

// ---------------------------------------------------------------------------
// Tensor Data Mover: DMA a 64-row x 32-elem bf16 tile (row stride
// `rowStride` elements) from global memory into contiguous LDS.
// D# per CDNA5 ISA ch.8: group0 = {count/lds_addr/global_addr/type},
// group1 = {data_size, tensor dims, tile dims, dim0 stride}.
// ---------------------------------------------------------------------------
__device__ __forceinline__ void tdm_load_64x32_bf16(unsigned long long gaddr,
                                                    unsigned ldsAddr,
                                                    unsigned rowStride)
{
  tdm_v4u g0;
  g0[0] = 1u;                                            // count=1 (valid D#)
  g0[1] = ldsAddr;                                       // LDS byte address
  g0[2] = (unsigned)(gaddr & 0xffffffffu);               // global_addr[31:0]
  g0[3] = ((unsigned)((gaddr >> 32) & 0x01ffffffu))      // global_addr[56:32]
        | (2u << 30);                                    // type = 2 ("image")
  tdm_v8i g1;
  g1[0] = (int)(1u << 16);                 // data_size = 1 -> 2 bytes
  g1[1] = (int)(rowStride << 16);          // tensor_dim0[15:0] @ bits 63:48
  g1[2] = (int)(512u << 16);               // tensor_dim1[15:0] @ bits 95:80
  g1[3] = (int)(32u << 16);                // tile_dim0 = 32    @ bits 127:112
  g1[4] = 64;                              // tile_dim1 = 64    @ bits 143:128
  g1[5] = (int)rowStride;                  // tensor_dim0_stride[31:0]
  g1[6] = 0;
  g1[7] = 0;
  tdm_v4i gz = {0, 0, 0, 0};
#if defined(__clang_major__) && (__clang_major__ >= 23)
  tdm_v8i gz8 = {0, 0, 0, 0, 0, 0, 0, 0};
  __builtin_amdgcn_tensor_load_to_lds(g0, g1, gz, gz, gz8, 0);
#else
  __builtin_amdgcn_tensor_load_to_lds(g0, g1, gz, gz, 0);
#endif
}

// ---------------------------------------------------------------------------
// Generic bf16-WMMA GEMM:  C = actOut( actIn(A) @ B + bias + residual )
//   A: M x K (lda), B: K x N (ldb), C: M x N (ldc), f32 in global memory.
//   ACTIN: 0=none, 1=relu, 2=layernorm (per-row stats + gamma/beta over K)
//   batched over blockIdx.z with element strides sA/sB/sC/sR.
// Block 256 = 8 waves; block tile 64x64; each wave: 2 stacked 16x16 tiles.
// Fast staging path (float4 = global_load_b128) when tiles are full/aligned.
// ---------------------------------------------------------------------------
template<int ACTIN, int ACTOUT, bool BIAS, bool RES>
__global__ __launch_bounds__(256)
void gemm_wmma(const float* __restrict__ A, const float* __restrict__ B,
               float* __restrict__ C,
               int M, int N, int K, int lda, int ldb, int ldc,
               long long sA, long long sB, long long sC,
               const float* __restrict__ bias,
               const float* __restrict__ res, int ldres, long long sR,
               const float* __restrict__ lnStats,
               const float* __restrict__ lnG, const float* __restrict__ lnB)
{
  __shared__ __bf16 As[64][36];
  __shared__ __bf16 Bs[32][68];
  const int bz = blockIdx.z;
  A += sA * bz; B += sB * bz; C += sC * bz;
  const int m0 = blockIdx.y * 64, n0 = blockIdx.x * 64;
  const int tid = threadIdx.x, wave = tid >> 5, lane = tid & 31;
  const int wm = (wave & 1) * 32, wn = (wave >> 1) * 16;
  const int rrow = lane & 15, g = lane >> 4;
  const bool fastA = ((lda & 3) == 0) && (m0 + 64 <= M) && ((K & 31) == 0);
  const bool fastB = ((ldb & 3) == 0) && (n0 + 64 <= N) && ((K & 31) == 0);
  v8f acc[2] = {};

  for (int k0 = 0; k0 < K; k0 += 32) {
    // ---- stage A tile (64x32) as bf16, applying input activation ----
    if (fastA) {
      #pragma unroll
      for (int it = 0; it < 2; it++) {
        int idx4 = tid + it * 256;               // 512 float4 = 2048 elems
        int r = idx4 >> 3, c4 = (idx4 & 7) * 4;
        const float4 vv = *(const float4*)(A + (long long)(m0 + r) * lda + k0 + c4);
        float x0 = vv.x, x1 = vv.y, x2 = vv.z, x3 = vv.w;
        if constexpr (ACTIN == 1) {
          x0 = fmaxf(x0, 0.f); x1 = fmaxf(x1, 0.f);
          x2 = fmaxf(x2, 0.f); x3 = fmaxf(x3, 0.f);
        } else if constexpr (ACTIN == 2) {
          float mu = lnStats[2ll * (m0 + r)], rs = lnStats[2ll * (m0 + r) + 1];
          const float* gp = lnG + k0 + c4;
          const float* bp = lnB + k0 + c4;
          x0 = (x0 - mu) * rs * gp[0] + bp[0];
          x1 = (x1 - mu) * rs * gp[1] + bp[1];
          x2 = (x2 - mu) * rs * gp[2] + bp[2];
          x3 = (x3 - mu) * rs * gp[3] + bp[3];
        }
        As[r][c4 + 0] = f2bf(x0); As[r][c4 + 1] = f2bf(x1);
        As[r][c4 + 2] = f2bf(x2); As[r][c4 + 3] = f2bf(x3);
      }
    } else {
      for (int idx = tid; idx < 64 * 32; idx += 256) {
        int r = idx >> 5, c = idx & 31;
        int gm = m0 + r, gk = k0 + c;
        float v = 0.f;
        if (gm < M && gk < K) {
          v = A[(long long)gm * lda + gk];
          if constexpr (ACTIN == 1) v = fmaxf(v, 0.f);
          else if constexpr (ACTIN == 2) {
            float mu = lnStats[2ll * gm], rs = lnStats[2ll * gm + 1];
            v = (v - mu) * rs * lnG[gk] + lnB[gk];
          }
        }
        As[r][c] = f2bf(v);
      }
    }
    // ---- stage B tile (32x64) ----
    if (fastB) {
      #pragma unroll
      for (int it = 0; it < 2; it++) {
        int idx4 = tid + it * 256;               // 512 float4 = 2048 elems
        int r = idx4 >> 4, c4 = (idx4 & 15) * 4;
        const float4 vv = *(const float4*)(B + (long long)(k0 + r) * ldb + n0 + c4);
        Bs[r][c4 + 0] = f2bf(vv.x); Bs[r][c4 + 1] = f2bf(vv.y);
        Bs[r][c4 + 2] = f2bf(vv.z); Bs[r][c4 + 3] = f2bf(vv.w);
      }
    } else {
      for (int idx = tid; idx < 32 * 64; idx += 256) {
        int r = idx >> 6, c = idx & 63;
        int gk = k0 + r, gn = n0 + c;
        float v = (gk < K && gn < N) ? B[(long long)gk * ldb + gn] : 0.f;
        Bs[r][c] = f2bf(v);
      }
    }
    __syncthreads();
    v16bf bf;
    #pragma unroll
    for (int e = 0; e < 16; e++) bf[e] = Bs[kmap(e, g)][wn + rrow];
    #pragma unroll
    for (int sub = 0; sub < 2; sub++) {
      v16bf af;
      #pragma unroll
      for (int e = 0; e < 16; e++) af[e] = As[wm + sub * 16 + rrow][kmap(e, g)];
      acc[sub] = wmma_bf16(af, bf, acc[sub]);
    }
    __syncthreads();
  }

  int gn = n0 + wn + rrow;
  if (gn < N) {
    float bv = 0.f;
    if constexpr (BIAS) bv = bias[gn];
    #pragma unroll
    for (int sub = 0; sub < 2; sub++) {
      #pragma unroll
      for (int r = 0; r < 8; r++) {
        int gm = m0 + wm + sub * 16 + r + 8 * g;
        if (gm < M) {
          float v = acc[sub][r] + bv;
          if constexpr (RES) v += (res + sR * bz)[(long long)gm * ldres + gn];
          if constexpr (ACTOUT == 1) v = fmaxf(v, 0.f);
          C[(long long)gm * ldc + gn] = v;
        }
      }
    }
  }
}

// ---------------------------------------------------------------------------
// Fused IPA logits: one K=32 WMMA per tile.
//   qf/kf: [res][head][32] packed bf16 features (q-scaled / rotated points),
//   DMA'd into LDS by the Tensor Data Mover (no VALU staging).
//   logits = qf.kf + WL*bias2d[i,j,h] - 0.5*WL*WC*gamma_h*(qq_i + kk_j)
// ---------------------------------------------------------------------------
__global__ __launch_bounds__(256)
void ipa_logits(const unsigned short* __restrict__ qf,
                const unsigned short* __restrict__ kf,
                const float* __restrict__ bias2d, const float* __restrict__ qq,
                const float* __restrict__ kk, const float* __restrict__ gsp,
                float* __restrict__ att)
{
  __shared__ __bf16 As[64][32];   // contiguous: filled by TDM
  __shared__ __bf16 Bs[64][32];
  const int h  = blockIdx.z;
  const int i0 = blockIdx.y * 64, j0 = blockIdx.x * 64;
  const int tid = threadIdx.x, wave = tid >> 5, lane = tid & 31;
  const int wm = (wave & 1) * 32, wn = (wave >> 1) * 16;
  const int rrow = lane & 15, g = lane >> 4;

  if (tid == 0) {
    unsigned long long qa = (unsigned long long)(uintptr_t)qf
                          + 2ull * ((unsigned long long)i0 * (H_ * 32) + h * 32);
    unsigned long long ka = (unsigned long long)(uintptr_t)kf
                          + 2ull * ((unsigned long long)j0 * (H_ * 32) + h * 32);
    tdm_load_64x32_bf16(qa, (unsigned)(uintptr_t)(&As[0][0]), H_ * 32);
    tdm_load_64x32_bf16(ka, (unsigned)(uintptr_t)(&Bs[0][0]), H_ * 32);
  }
  __builtin_amdgcn_s_wait_tensorcnt(0);
  __syncthreads();

  v8f acc[2] = {};
  v16bf bf;          // kf tile is [j][k]: read "A-style" = transposed B
  #pragma unroll
  for (int e = 0; e < 16; e++) bf[e] = Bs[wn + rrow][kmap(e, g)];
  #pragma unroll
  for (int sub = 0; sub < 2; sub++) {
    v16bf af;
    #pragma unroll
    for (int e = 0; e < 16; e++) af[e] = As[wm + sub * 16 + rrow][kmap(e, g)];
    acc[sub] = wmma_bf16(af, bf, acc[sub]);
  }

  const float hg = 0.5f * WL_ * WC_ * gsp[h];
  const int gj = j0 + wn + rrow;
  const float kkv = kk[gj * H_ + h] * hg;
  #pragma unroll
  for (int sub = 0; sub < 2; sub++) {
    #pragma unroll
    for (int r = 0; r < 8; r++) {
      int gi = i0 + wm + sub * 16 + r + 8 * g;
      float v = acc[sub][r]
              + WL_ * bias2d[((long long)gi * NRES + gj) * H_ + h]
              - hg * qq[gi * H_ + h] - kkv;
      att[((long long)h * NRES + gi) * NRES + gj] = v;
    }
  }
}

// ---------------------------------------------------------------------------
// opair[i,h,:] = sum_j att[h,i,j] * LN(z[i,j,:])   (M=16pad, N=64, K=512)
// 4 waves; LN of z applied while staging into LDS; writes into cat slice.
// ---------------------------------------------------------------------------
__global__ __launch_bounds__(128)
void ipa_opair(const float* __restrict__ att, const float* __restrict__ z,
               const float* __restrict__ zst, const float* __restrict__ zg,
               const float* __restrict__ zb, float* __restrict__ cat)
{
  __shared__ __bf16 As[16][36];
  __shared__ __bf16 Bs[32][68];
  const int i  = blockIdx.y;
  const int n0 = blockIdx.x * 64;
  const int tid = threadIdx.x, wave = tid >> 5, lane = tid & 31;
  const int wn = wave * 16, rrow = lane & 15, g = lane >> 4;
  v8f acc = {};

  for (int k0 = 0; k0 < NRES; k0 += 32) {
    // A tile: 16x32 (12 live head rows), rows contiguous in att
    for (int idx4 = tid; idx4 < 128; idx4 += 128) {
      int r = idx4 >> 3, c4 = (idx4 & 7) * 4;
      float x0 = 0.f, x1 = 0.f, x2 = 0.f, x3 = 0.f;
      if (r < H_) {
        const float4 vv = *(const float4*)(att + ((long long)r * NRES + i) * NRES + k0 + c4);
        x0 = vv.x; x1 = vv.y; x2 = vv.z; x3 = vv.w;
      }
      As[r][c4 + 0] = f2bf(x0); As[r][c4 + 1] = f2bf(x1);
      As[r][c4 + 2] = f2bf(x2); As[r][c4 + 3] = f2bf(x3);
    }
    // B tile: 32x64 slice of LN(z[i]) -- rows always full & aligned
    #pragma unroll
    for (int it = 0; it < 4; it++) {
      int idx4 = tid + it * 128;                  // 512 float4 = 2048 elems
      int r = idx4 >> 4, c4 = (idx4 & 15) * 4;
      long long jrow = (long long)i * NRES + k0 + r;
      float mu = zst[2 * jrow], rs = zst[2 * jrow + 1];
      const float4 vv = *(const float4*)(z + jrow * CZ_ + n0 + c4);
      const float4 gg = *(const float4*)(zg + n0 + c4);
      const float4 bb = *(const float4*)(zb + n0 + c4);
      Bs[r][c4 + 0] = f2bf((vv.x - mu) * rs * gg.x + bb.x);
      Bs[r][c4 + 1] = f2bf((vv.y - mu) * rs * gg.y + bb.y);
      Bs[r][c4 + 2] = f2bf((vv.z - mu) * rs * gg.z + bb.z);
      Bs[r][c4 + 3] = f2bf((vv.w - mu) * rs * gg.w + bb.w);
    }
    __syncthreads();
    v16bf af, bf;
    #pragma unroll
    for (int e = 0; e < 16; e++) bf[e] = Bs[kmap(e, g)][wn + rrow];
    #pragma unroll
    for (int e = 0; e < 16; e++) af[e] = As[rrow][kmap(e, g)];
    acc = wmma_bf16(af, bf, acc);
    __syncthreads();
  }
  int gn = n0 + wn + rrow;
  #pragma unroll
  for (int r = 0; r < 8; r++) {
    int hh = r + 8 * g;
    if (hh < H_)
      cat[(long long)i * 2208 + 576 + hh * CZ_ + gn] = acc[r];
  }
}

// ---------------------------------------------------------------------------
// Small VALU kernels
// ---------------------------------------------------------------------------
__global__ __launch_bounds__(128)
void ln384_kernel(const float* __restrict__ in, float* __restrict__ out,
                  const float* __restrict__ g, const float* __restrict__ b)
{
  long long row = blockIdx.x;
  const float* p = in + row * CS_;
  int tid = threadIdx.x;
  float x0 = p[tid], x1 = p[tid + 128], x2 = p[tid + 256];
  __shared__ float red[128];
  red[tid] = x0 + x1 + x2; __syncthreads();
  for (int s = 64; s > 0; s >>= 1) { if (tid < s) red[tid] += red[tid + s]; __syncthreads(); }
  float mu = red[0] * (1.0f / CS_); __syncthreads();
  float d0 = x0 - mu, d1 = x1 - mu, d2 = x2 - mu;
  red[tid] = d0 * d0 + d1 * d1 + d2 * d2; __syncthreads();
  for (int s = 64; s > 0; s >>= 1) { if (tid < s) red[tid] += red[tid + s]; __syncthreads(); }
  float rs = rsqrtf(red[0] * (1.0f / CS_) + 1e-5f);
  float* q = out + row * CS_;
  q[tid]       = d0 * rs * g[tid]       + b[tid];
  q[tid + 128] = d1 * rs * g[tid + 128] + b[tid + 128];
  q[tid + 256] = d2 * rs * g[tid + 256] + b[tid + 256];
}

__global__ __launch_bounds__(128)
void zstats_kernel(const float* __restrict__ z, float* __restrict__ st)
{
  long long row = blockIdx.x;
  const float* p = z + row * CZ_;
  int tid = threadIdx.x;
  float x = p[tid];
  __shared__ float red[128];
  red[tid] = x; __syncthreads();
  for (int s = 64; s > 0; s >>= 1) { if (tid < s) red[tid] += red[tid + s]; __syncthreads(); }
  float mu = red[0] * (1.0f / CZ_); __syncthreads();
  float d = x - mu; red[tid] = d * d; __syncthreads();
  for (int s = 64; s > 0; s >>= 1) { if (tid < s) red[tid] += red[tid + s]; __syncthreads(); }
  if (tid == 0) { st[2 * row] = mu; st[2 * row + 1] = rsqrtf(red[0] * (1.0f / CZ_) + 1e-5f); }
}

__global__ __launch_bounds__(256)
void softmax512_kernel(float* __restrict__ a)
{
  long long row = blockIdx.x;
  float* p = a + row * NRES;
  int tid = threadIdx.x;
  float x0 = p[tid], x1 = p[tid + 256];
  __shared__ float red[256];
  red[tid] = fmaxf(x0, x1); __syncthreads();
  for (int s = 128; s > 0; s >>= 1) { if (tid < s) red[tid] = fmaxf(red[tid], red[tid + s]); __syncthreads(); }
  float m = red[0]; __syncthreads();
  float e0 = expf(x0 - m), e1 = expf(x1 - m);
  red[tid] = e0 + e1; __syncthreads();
  for (int s = 128; s > 0; s >>= 1) { if (tid < s) red[tid] += red[tid + s]; __syncthreads(); }
  float inv = 1.0f / red[0];
  p[tid] = e0 * inv; p[tid + 256] = e1 * inv;
}

__global__ void softplus12_kernel(const float* __restrict__ g, float* __restrict__ o)
{
  int i = threadIdx.x;
  if (i < H_) { float x = g[i]; o[i] = (x > 20.f) ? x : log1pf(expf(x)); }
}

__global__ void init_rt_kernel(float* __restrict__ R, float* __restrict__ t)
{
  int i = blockIdx.x * blockDim.x + threadIdx.x;
  if (i < NRES) {
    float* r = R + i * 9;
    r[0]=1.f; r[1]=0.f; r[2]=0.f; r[3]=0.f; r[4]=1.f; r[5]=0.f; r[6]=0.f; r[7]=0.f; r[8]=1.f;
    t[i*3+0]=0.f; t[i*3+1]=0.f; t[i*3+2]=0.f;
  }
}

// pack per-(i,h) fused attention features (bf16) + point sums of squares
__global__ __launch_bounds__(64)
void ipa_pack(const float* __restrict__ q, const float* __restrict__ kx,
              const float* __restrict__ qp, const float* __restrict__ kp,
              const float* __restrict__ R, const float* __restrict__ t,
              const float* __restrict__ gsp,
              unsigned short* __restrict__ qf, unsigned short* __restrict__ kf,
              float* __restrict__ qq, float* __restrict__ kk)
{
  int i = blockIdx.x, h = threadIdx.x;
  if (h >= H_) return;
  float Rm[9], tv[3];
  #pragma unroll
  for (int j = 0; j < 9; j++) Rm[j] = R[i * 9 + j];
  #pragma unroll
  for (int c = 0; c < 3; c++) tv[c] = t[i * 3 + c];
  unsigned short* qo = qf + ((long long)i * H_ + h) * 32;
  unsigned short* ko = kf + ((long long)i * H_ + h) * 32;
  float Ah = WL_ * WC_ * gsp[h];
  #pragma unroll
  for (int c = 0; c < 16; c++) {
    qo[c] = f2bfu(q[i * 192 + h * 16 + c] * (WL_ * 0.25f));
    ko[c] = f2bfu(kx[i * 192 + h * 16 + c]);
  }
  float sq = 0.f, sk = 0.f;
  #pragma unroll
  for (int p = 0; p < 4; p++) {
    const float* a = qp + (long long)i * 144 + h * 12 + p * 3;
    float gx = Rm[0]*a[0]+Rm[1]*a[1]+Rm[2]*a[2]+tv[0];
    float gy = Rm[3]*a[0]+Rm[4]*a[1]+Rm[5]*a[2]+tv[1];
    float gz = Rm[6]*a[0]+Rm[7]*a[1]+Rm[8]*a[2]+tv[2];
    sq += gx*gx + gy*gy + gz*gz;
    qo[16+p*3+0]=f2bfu(gx*Ah); qo[16+p*3+1]=f2bfu(gy*Ah); qo[16+p*3+2]=f2bfu(gz*Ah);
    const float* bp = kp + (long long)i * 144 + h * 12 + p * 3;
    float hx = Rm[0]*bp[0]+Rm[1]*bp[1]+Rm[2]*bp[2]+tv[0];
    float hy = Rm[3]*bp[0]+Rm[4]*bp[1]+Rm[5]*bp[2]+tv[1];
    float hz = Rm[6]*bp[0]+Rm[7]*bp[1]+Rm[8]*bp[2]+tv[2];
    sk += hx*hx + hy*hy + hz*hz;
    ko[16+p*3+0]=f2bfu(hx); ko[16+p*3+1]=f2bfu(hy); ko[16+p*3+2]=f2bfu(hz);
  }
  #pragma unroll
  for (int c = 28; c < 32; c++) { qo[c] = 0; ko[c] = 0; }
  qq[i * H_ + h] = sq; kk[i * H_ + h] = sk;
}

__global__ __launch_bounds__(96)
void ipa_vrot(const float* __restrict__ vp, const float* __restrict__ R,
              const float* __restrict__ t, float* __restrict__ vpg)
{
  int i = blockIdx.x, tid = threadIdx.x;
  if (tid >= 96) return;
  int h = tid / 8, p = tid % 8;
  float Rm[9], tv[3];
  #pragma unroll
  for (int j = 0; j < 9; j++) Rm[j] = R[i * 9 + j];
  #pragma unroll
  for (int c = 0; c < 3; c++) tv[c] = t[i * 3 + c];
  const float* s = vp + (long long)i * 288 + h * 24 + p * 3;
  float gx = Rm[0]*s[0]+Rm[1]*s[1]+Rm[2]*s[2]+tv[0];
  float gy = Rm[3]*s[0]+Rm[4]*s[1]+Rm[5]*s[2]+tv[1];
  float gz = Rm[6]*s[0]+Rm[7]*s[1]+Rm[8]*s[2]+tv[2];
  float* o = vpg + (long long)i * 288 + h * 24 + p * 3;
  o[0]=gx; o[1]=gy; o[2]=gz;
}

// op = R^T (opg - t), onorm -> into cat slices
__global__ __launch_bounds__(96)
void ipa_oprot(const float* __restrict__ opg, const float* __restrict__ R,
               const float* __restrict__ t, float* __restrict__ cat)
{
  int i = blockIdx.x, tid = threadIdx.x;
  if (tid >= 96) return;
  int h = tid / 8, p = tid % 8;
  float Rm[9], tv[3];
  #pragma unroll
  for (int j = 0; j < 9; j++) Rm[j] = R[i * 9 + j];
  #pragma unroll
  for (int c = 0; c < 3; c++) tv[c] = t[i * 3 + c];
  const float* s = opg + (long long)i * 288 + h * 24 + p * 3;
  float dx = s[0] - tv[0], dy = s[1] - tv[1], dz = s[2] - tv[2];
  float ox = Rm[0]*dx + Rm[3]*dy + Rm[6]*dz;
  float oy = Rm[1]*dx + Rm[4]*dy + Rm[7]*dz;
  float oz = Rm[2]*dx + Rm[5]*dy + Rm[8]*dz;
  float* cp = cat + (long long)i * 2208;
  int pi = h * 8 + p;
  cp[192 + pi*3 + 0] = ox; cp[192 + pi*3 + 1] = oy; cp[192 + pi*3 + 2] = oz;
  cp[480 + pi] = sqrtf(ox*ox + oy*oy + oz*oz + 1e-8f);
}

__global__ __launch_bounds__(256)
void frame_update(const float* __restrict__ o6, float* __restrict__ R,
                  float* __restrict__ t, float* __restrict__ fr)
{
  int i = blockIdx.x * 256 + threadIdx.x;
  if (i >= NRES) return;
  const float* o = o6 + i * 6;
  float bx = o[0], by = o[1], bz = o[2];
  float inv = rsqrtf(1.f + bx*bx + by*by + bz*bz);
  float w = inv, x = bx*inv, y = by*inv, z = bz*inv;
  float Rn[9];
  Rn[0]=1.f-2.f*(y*y+z*z); Rn[1]=2.f*(x*y-w*z);     Rn[2]=2.f*(x*z+w*y);
  Rn[3]=2.f*(x*y+w*z);     Rn[4]=1.f-2.f*(x*x+z*z); Rn[5]=2.f*(y*z-w*x);
  Rn[6]=2.f*(x*z-w*y);     Rn[7]=2.f*(y*z+w*x);     Rn[8]=1.f-2.f*(x*x+y*y);
  float Rm[9], tv[3], tn[3] = { o[3], o[4], o[5] };
  #pragma unroll
  for (int j = 0; j < 9; j++) Rm[j] = R[i * 9 + j];
  #pragma unroll
  for (int c = 0; c < 3; c++) tv[c] = t[i * 3 + c];
  float tnew[3], Rnew[9];
  #pragma unroll
  for (int r = 0; r < 3; r++) {
    tnew[r] = Rm[r*3+0]*tn[0] + Rm[r*3+1]*tn[1] + Rm[r*3+2]*tn[2] + tv[r];
    #pragma unroll
    for (int c = 0; c < 3; c++)
      Rnew[r*3+c] = Rm[r*3+0]*Rn[0*3+c] + Rm[r*3+1]*Rn[1*3+c] + Rm[r*3+2]*Rn[2*3+c];
  }
  #pragma unroll
  for (int j = 0; j < 9; j++) R[i * 9 + j] = Rnew[j];
  #pragma unroll
  for (int c = 0; c < 3; c++) t[i * 3 + c] = tnew[c];
  float* F = fr + (long long)i * 16;
  #pragma unroll
  for (int r = 0; r < 3; r++) {
    F[r*4+0]=Rnew[r*3+0]; F[r*4+1]=Rnew[r*3+1]; F[r*4+2]=Rnew[r*3+2]; F[r*4+3]=tnew[r];
  }
  F[12]=0.f; F[13]=0.f; F[14]=0.f; F[15]=1.f;
}

// ---------------------------------------------------------------------------
// Host orchestration
// ---------------------------------------------------------------------------
static inline void gemm(hipStream_t st,
                        const float* A, const float* B, float* Cp,
                        int M, int N, int K, int lda, int ldb, int ldc,
                        const float* bias, const float* res, int ldres,
                        int actIn, int actOut,
                        const float* lnStats = nullptr,
                        const float* lnG = nullptr, const float* lnB = nullptr,
                        int batch = 1, long long sA = 0, long long sB = 0,
                        long long sC = 0, long long sR = 0)
{
  dim3 grid((N + 63) / 64, (M + 63) / 64, batch);
#define GEMM_ARGS A, B, Cp, M, N, K, lda, ldb, ldc, sA, sB, sC, bias, res, ldres, sR, lnStats, lnG, lnB
  if (actIn == 2)                    gemm_wmma<2,0,true, false><<<grid,256,0,st>>>(GEMM_ARGS);
  else if (actIn == 1 && actOut == 1) gemm_wmma<1,1,true, false><<<grid,256,0,st>>>(GEMM_ARGS);
  else if (actIn == 1 && res)        gemm_wmma<1,0,true, true ><<<grid,256,0,st>>>(GEMM_ARGS);
  else if (actIn == 1)               gemm_wmma<1,0,true, false><<<grid,256,0,st>>>(GEMM_ARGS);
  else if (actOut == 1)              gemm_wmma<0,1,true, false><<<grid,256,0,st>>>(GEMM_ARGS);
  else if (res)                      gemm_wmma<0,0,true, true ><<<grid,256,0,st>>>(GEMM_ARGS);
  else if (bias)                     gemm_wmma<0,0,true, false><<<grid,256,0,st>>>(GEMM_ARGS);
  else                               gemm_wmma<0,0,false,false><<<grid,256,0,st>>>(GEMM_ARGS);
#undef GEMM_ARGS
}

extern "C" void kernel_launch(void* const* d_in, const int* in_sizes, int n_in,
                              void* d_out_v, int out_size, void* d_ws, size_t ws_size,
                              hipStream_t stream)
{
  (void)in_sizes; (void)n_in; (void)out_size; (void)ws_size;
  const float* s_in = (const float*)d_in[0];
  const float* z_in = (const float*)d_in[1];
  // d_in[2] = F (unused by the reference computation)
  auto P = [&](int i) { return (const float*)d_in[3 + i]; };
  const float *ln_s_g = P(0), *ln_s_b = P(1), *ln_z_g = P(2), *ln_z_b = P(3);
  const float *lin_in_w = P(4), *lin_in_b = P(5);
  const float *wq = P(6), *bq = P(7), *wk = P(8), *bk = P(9), *wv = P(10), *bv = P(11);
  const float *wqp = P(12), *bqp = P(13), *wkp = P(14), *bkp = P(15), *wvp = P(16), *bvp = P(17);
  const float *wbp = P(18), *bbp = P(19), *gamma = P(20);
  const float *wout = P(21), *bout = P(22);
  const float *ln_ipa_g = P(23), *ln_ipa_b = P(24);
  const float *tr_w1 = P(25), *tr_b1 = P(26), *tr_w2 = P(27), *tr_b2 = P(28);
  const float *tr_w3 = P(29), *tr_b3 = P(30), *tr_ln_g = P(31), *tr_ln_b = P(32);
  const float *bb_w = P(33), *bb_b = P(34);
  const float *ar_win = P(35), *ar_bin = P(36), *ar_winit = P(37), *ar_binit = P(38);
  const float *ar_wout = P(39), *ar_bout = P(40);
  const float *arw1[2] = { P(41), P(45) }, *arb1[2] = { P(42), P(46) };
  const float *arw2[2] = { P(43), P(47) }, *arb2[2] = { P(44), P(48) };

  float* wsp = (float*)d_ws;
  size_t off = 0;
  auto alloc = [&](size_t n) { float* p = wsp + off; off += n; return p; };
  float* sinit = alloc(NRES * CS_);
  float* scur  = alloc(NRES * CS_);
  float* tmpA  = alloc(NRES * CS_);
  float* tmpB  = alloc(NRES * CS_);
  float* zst   = alloc(2ll * NRES * NRES);
  float* b2d   = alloc((size_t)NRES * NRES * H_);
  float* qb    = alloc(NRES * 192);
  float* kb    = alloc(NRES * 192);
  float* vb    = alloc(NRES * 192);
  float* qp    = alloc(NRES * 144);
  float* kp    = alloc(NRES * 144);
  float* vp    = alloc(NRES * 288);
  float* vpg   = alloc(NRES * 288);
  unsigned short* qf = (unsigned short*)alloc(NRES * H_ * 16);  // bf16 storage
  unsigned short* kf = (unsigned short*)alloc(NRES * H_ * 16);
  float* qq    = alloc(NRES * H_);
  float* kk    = alloc(NRES * H_);
  float* att   = alloc((size_t)H_ * NRES * NRES);
  float* opg   = alloc(NRES * 288);
  float* cat   = alloc(NRES * 2208);
  float* o6    = alloc(NRES * 6);
  float* Rbuf  = alloc(NRES * 9);
  float* tbuf  = alloc(NRES * 3);
  float* gsp   = alloc(16);
  float* aA    = alloc(NRES * C_);
  float* aH    = alloc(NRES * C_);

  float* frames_out = (float*)d_out_v;                     // [8][512][4][4]
  float* angles_out = frames_out + 8ll * NRES * 16;        // [8][512][7][2]

  // ----- loop-invariant preprocessing -----
  ln384_kernel<<<NRES, 128, 0, stream>>>(s_in, sinit, ln_s_g, ln_s_b);
  zstats_kernel<<<NRES * NRES, 128, 0, stream>>>(z_in, zst);
  softplus12_kernel<<<1, 32, 0, stream>>>(gamma, gsp);
  init_rt_kernel<<<2, 256, 0, stream>>>(Rbuf, tbuf);
  // s = sinit @ lin_in_w + b
  gemm(stream, sinit, lin_in_w, scur, NRES, CS_, CS_, CS_, CS_, CS_,
       lin_in_b, nullptr, 0, 0, 0);
  // bias2d = LN(z) @ wb + bb  (z loop-invariant -> computed once, stays in L2)
  gemm(stream, z_in, wbp, b2d, NRES * NRES, H_, CZ_, CZ_, H_, H_,
       bbp, nullptr, 0, 2, 0, zst, ln_z_g, ln_z_b);

  for (int layer = 0; layer < 8; layer++) {
    // projections
    gemm(stream, scur, wq,  qb, NRES, 192, CS_, CS_, 192, 192, bq,  nullptr, 0, 0, 0);
    gemm(stream, scur, wk,  kb, NRES, 192, CS_, CS_, 192, 192, bk,  nullptr, 0, 0, 0);
    gemm(stream, scur, wv,  vb, NRES, 192, CS_, CS_, 192, 192, bv,  nullptr, 0, 0, 0);
    gemm(stream, scur, wqp, qp, NRES, 144, CS_, CS_, 144, 144, bqp, nullptr, 0, 0, 0);
    gemm(stream, scur, wkp, kp, NRES, 144, CS_, CS_, 144, 144, bkp, nullptr, 0, 0, 0);
    gemm(stream, scur, wvp, vp, NRES, 288, CS_, CS_, 288, 288, bvp, nullptr, 0, 0, 0);
    // rotate points + pack fused bf16 features
    ipa_pack<<<NRES, 64, 0, stream>>>(qb, kb, qp, kp, Rbuf, tbuf, gsp, qf, kf, qq, kk);
    ipa_vrot<<<NRES, 96, 0, stream>>>(vp, Rbuf, tbuf, vpg);
    // fused logits (QK + point-distance + pair bias) via TDM + WMMA, softmax
    ipa_logits<<<dim3(8, 8, H_), 256, 0, stream>>>(qf, kf, b2d, qq, kk, gsp, att);
    softmax512_kernel<<<H_ * NRES, 256, 0, stream>>>(att);
    // o = a @ v  (per head, written straight into cat[:,0:192])
    gemm(stream, att, vb, cat, NRES, 16, NRES, NRES, 192, 2208,
         nullptr, nullptr, 0, 0, 0, nullptr, nullptr, nullptr,
         H_, (long long)NRES * NRES, 16, 16, 0);
    // opg = a @ vpg (per head)
    gemm(stream, att, vpg, opg, NRES, 24, NRES, NRES, 288, 288,
         nullptr, nullptr, 0, 0, 0, nullptr, nullptr, nullptr,
         H_, (long long)NRES * NRES, 24, 24, 0);
    // opair into cat[:,576:2208]; op/onorm into cat[:,192:576]
    ipa_opair<<<dim3(2, NRES), 128, 0, stream>>>(att, z_in, zst, ln_z_g, ln_z_b, cat);
    ipa_oprot<<<NRES, 96, 0, stream>>>(opg, Rbuf, tbuf, cat);
    // s = s + cat @ wout + bout ; LN
    gemm(stream, cat, wout, scur, NRES, CS_, 2208, 2208, CS_, CS_,
         bout, scur, CS_, 0, 0);
    ln384_kernel<<<NRES, 128, 0, stream>>>(scur, scur, ln_ipa_g, ln_ipa_b);
    // transition MLP + LN
    gemm(stream, scur, tr_w1, tmpA, NRES, CS_, CS_, CS_, CS_, CS_, tr_b1, nullptr, 0, 0, 1);
    gemm(stream, tmpA, tr_w2, tmpB, NRES, CS_, CS_, CS_, CS_, CS_, tr_b2, nullptr, 0, 0, 1);
    gemm(stream, tmpB, tr_w3, scur, NRES, CS_, CS_, CS_, CS_, CS_, tr_b3, scur, CS_, 0, 0);
    ln384_kernel<<<NRES, 128, 0, stream>>>(scur, scur, tr_ln_g, tr_ln_b);
    // backbone update -> frames[layer]
    gemm(stream, scur, bb_w, o6, NRES, 6, CS_, CS_, 6, 6, bb_b, nullptr, 0, 0, 0);
    frame_update<<<2, 256, 0, stream>>>(o6, Rbuf, tbuf, frames_out + (long long)layer * NRES * 16);
    // angle resnet -> angles[layer]
    gemm(stream, scur,  ar_win,   aA, NRES, C_, CS_, CS_, C_, C_, ar_bin,   nullptr, 0, 1, 0);
    gemm(stream, sinit, ar_winit, aA, NRES, C_, CS_, CS_, C_, C_, ar_binit, aA, C_,  1, 0);
    for (int l = 0; l < 2; l++) {
      gemm(stream, aA, arw1[l], aH, NRES, C_, C_, C_, C_, C_, arb1[l], nullptr, 0, 1, 1);
      gemm(stream, aH, arw2[l], aA, NRES, C_, C_, C_, C_, C_, arb2[l], aA, C_,   0, 0);
    }
    gemm(stream, aA, ar_wout, angles_out + (long long)layer * NRES * 14,
         NRES, 2 * NT_, C_, C_, 2 * NT_, 2 * NT_, ar_bout, nullptr, 0, 1, 0);
  }
}
